// PartialGNNBilinearHead_6923487281811
// MI455X (gfx1250) — matrix operand order repaired and checked by reading, not compiled, not run on hardware
//
#include <hip/hip_runtime.h>
#include <math.h>

typedef __bf16 bf16_t;
typedef __attribute__((ext_vector_type(16))) __bf16 v16bf;
typedef __attribute__((ext_vector_type(8)))  __bf16 v8bf;
typedef __attribute__((ext_vector_type(4)))  __bf16 v4bf;
typedef __attribute__((ext_vector_type(8)))  float  v8f;

#define BM 128
#define BN 64
#define BK 32
#define LDS_A_STRIDE 40   // elems; 80 B rows: ds_read_b128 stays 16B aligned, banks staggered
#define LDS_B_STRIDE 40

constexpr int EP_NONE  = 0;
constexpr int EP_BIAS  = 1;
constexpr int EP_GELU  = 2;
constexpr int EP_ACCUM = 4;

__device__ __forceinline__ float gelu_tanh(float x) {
    const float k0 = 0.7978845608028654f;   // sqrt(2/pi)
    const float k1 = 0.044715f;
    float x3 = x * x * x;
    return 0.5f * x * (1.0f + tanhf(k0 * (x + k1 * x3)));
}

__device__ __forceinline__ v4bf cvt4(float4 v) {
    v4bf w;
    w[0] = (bf16_t)v.x; w[1] = (bf16_t)v.y; w[2] = (bf16_t)v.z; w[3] = (bf16_t)v.w;
    return w;
}

// ---------------------------------------------------------------------------
// Generic WMMA GEMM: C[M,N] (+)= epilogue( A[M,K] * B + bias )
//   BT=false: B is [K,N] row-major.  BT=true: B is [N,K] row-major (NT gemm).
// Block: 256 threads (8 waves), tile 128x64xK32, each wave = 32x32 (2x2 WMMA).
// Double-buffered LDS pipeline: next tile's global_load_b128s issue before the
// current tile's WMMAs; regs->LDS after compute; one barrier per K-step.
// OOB rows/cols use clamped addresses (no zero-fill needed: a garbage A-row /
// B-col only affects the same-index output row/col, which is never stored).
// ---------------------------------------------------------------------------
template<bool BT, int EPI>
__global__ __launch_bounds__(256)
void wmma_gemm_kernel(const float* __restrict__ A,
                      const float* __restrict__ B,
                      const float* __restrict__ bias,
                      float* __restrict__ C,
                      int M, int N, int K)
{
    __shared__ __align__(16) bf16_t As[2][BM * LDS_A_STRIDE];
    __shared__ __align__(16) bf16_t Bs[2][BN * LDS_B_STRIDE];

    const int tid  = threadIdx.x;
    const int lane = tid & 31;
    const int wid  = tid >> 5;          // 0..7
    const int wm   = (wid >> 1) * 32;   // wave M offset: 0,32,64,96
    const int wn   = (wid & 1) * 32;    // wave N offset: 0,32
    const int hh   = lane >> 4;         // lane half (ISA fragment layout)
    const int ml   = lane & 15;

    const int m0 = blockIdx.x * BM;
    const int n0 = blockIdx.y * BN;

    // staging coordinates (all unconditional, clamped addresses)
    const int ar = tid >> 3;            // A row base 0..31 (+32*it)
    const int ac = (tid & 7) * 4;       // A col 0,4,..,28
    const int bnr = tid >> 3;           // BT: Bt row base 0..31 (+32*it)
    const int bnc = (tid & 7) * 4;
    const int bk  = tid >> 4;           // NN: k base 0..15 (+16*it)
    const int bn4 = (tid & 15) * 4;     // NN: n group 0,4,..,60

    float4 pa[4];
    float4 pb[2];

    auto load_tiles = [&](int k0) {
#pragma unroll
        for (int it = 0; it < 4; ++it) {
            int gr = m0 + ar + 32 * it;
            gr = (gr < M) ? gr : (M - 1);
            pa[it] = *(const float4*)(A + (size_t)gr * K + (k0 + ac));
        }
        if constexpr (BT) {
#pragma unroll
            for (int it = 0; it < 2; ++it) {
                int gn = n0 + bnr + 32 * it;
                gn = (gn < N) ? gn : (N - 1);
                pb[it] = *(const float4*)(B + (size_t)gn * K + (k0 + bnc));
            }
        } else {
#pragma unroll
            for (int it = 0; it < 2; ++it) {
                int gn = n0 + bn4;
                gn = (gn <= N - 4) ? gn : (N - 4);
                pb[it] = *(const float4*)(B + (size_t)(k0 + bk + 16 * it) * N + gn);
            }
        }
    };

    auto store_tiles = [&](int buf) {
#pragma unroll
        for (int it = 0; it < 4; ++it)
            *(v4bf*)(&As[buf][(ar + 32 * it) * LDS_A_STRIDE + ac]) = cvt4(pa[it]);
        if constexpr (BT) {
#pragma unroll
            for (int it = 0; it < 2; ++it)
                *(v4bf*)(&Bs[buf][(bnr + 32 * it) * LDS_B_STRIDE + bnc]) = cvt4(pb[it]);
        } else {
#pragma unroll
            for (int it = 0; it < 2; ++it) {
                int k = bk + 16 * it;
                Bs[buf][(bn4 + 0) * LDS_B_STRIDE + k] = (bf16_t)pb[it].x;
                Bs[buf][(bn4 + 1) * LDS_B_STRIDE + k] = (bf16_t)pb[it].y;
                Bs[buf][(bn4 + 2) * LDS_B_STRIDE + k] = (bf16_t)pb[it].z;
                Bs[buf][(bn4 + 3) * LDS_B_STRIDE + k] = (bf16_t)pb[it].w;
            }
        }
    };

    v8f acc[2][2];
#pragma unroll
    for (int a = 0; a < 2; ++a)
#pragma unroll
        for (int b = 0; b < 2; ++b)
#pragma unroll
            for (int e = 0; e < 8; ++e) acc[a][b][e] = 0.0f;

    const int nk = K / BK;
    load_tiles(0);
    store_tiles(0);

    for (int kt = 0; kt < nk; ++kt) {
        const int cur = kt & 1;
        __syncthreads();
        if (kt + 1 < nk) load_tiles((kt + 1) * BK);   // overlap with WMMA below

        // ---- fragment loads (ISA 7.12.2 layouts) + 4 WMMAs ----
        v16bf afrag[2], bfrag[2];
#pragma unroll
        for (int t = 0; t < 2; ++t) {
            // A 16x32: lane ml = row; elems 0..7 -> k=e+8*hh, 8..15 -> k=(e-8)+16+8*hh
            const bf16_t* ab = &As[cur][(wm + t * 16 + ml) * LDS_A_STRIDE];
            v8bf alo = *(const v8bf*)(ab + 8 * hh);
            v8bf ahi = *(const v8bf*)(ab + 16 + 8 * hh);
#pragma unroll
            for (int e = 0; e < 8; ++e) { afrag[t][e] = alo[e]; afrag[t][e + 8] = ahi[e]; }
            // B 32x16: lane ml = col; elems 0..7 -> k=e+8*hh, 8..15 -> k=(e-8)+16+8*hh
            const bf16_t* bbp = &Bs[cur][(wn + t * 16 + ml) * LDS_B_STRIDE];
            v8bf blo = *(const v8bf*)(bbp + 8 * hh);
            v8bf bhi = *(const v8bf*)(bbp + 16 + 8 * hh);
#pragma unroll
            for (int e = 0; e < 8; ++e) { bfrag[t][e] = blo[e]; bfrag[t][e + 8] = bhi[e]; }
        }
#pragma unroll
        for (int tm = 0; tm < 2; ++tm)
#pragma unroll
            for (int tn = 0; tn < 2; ++tn)
                acc[tm][tn] = __builtin_amdgcn_wmma_f32_16x16x32_bf16(
                    false, afrag[tm], false, bfrag[tn],
                    (short)0, acc[tm][tn], false, false);

        if (kt + 1 < nk) store_tiles(cur ^ 1);
    }

    // ---- epilogue: C/D layout: VGPR r -> row = r + 8*hh, col = ml ----
#pragma unroll
    for (int tm = 0; tm < 2; ++tm)
#pragma unroll
        for (int tn = 0; tn < 2; ++tn) {
            int gn = n0 + wn + tn * 16 + ml;
            if (gn >= N) continue;
            float bv = (EPI & EP_BIAS) ? bias[gn] : 0.0f;
#pragma unroll
            for (int r = 0; r < 8; ++r) {
                int gm = m0 + wm + tm * 16 + r + 8 * hh;
                if (gm >= M) continue;
                float v = acc[tm][tn][r] + bv;
                if (EPI & EP_GELU) v = gelu_tanh(v);
                size_t o = (size_t)gm * N + gn;
                if (EPI & EP_ACCUM) C[o] += v;
                else                C[o] = v;
            }
        }
}

// ---------------------------------------------------------------------------
// LayerNorm over last dim D (block per row).  RELU_RES: out = res + relu(ln(x))
// ---------------------------------------------------------------------------
template<bool RELU_RES>
__global__ __launch_bounds__(256)
void ln_kernel(const float* __restrict__ x, const float* __restrict__ g,
               const float* __restrict__ b, const float* __restrict__ res,
               float* __restrict__ out, int D)
{
    __shared__ float sbuf[256];
    const int row = blockIdx.x;
    const int tid = threadIdx.x;
    const float* xr = x + (size_t)row * D;

    float s = 0.0f;
    for (int i = tid; i < D; i += 256) s += xr[i];
    sbuf[tid] = s; __syncthreads();
    for (int st = 128; st > 0; st >>= 1) {
        if (tid < st) sbuf[tid] += sbuf[tid + st];
        __syncthreads();
    }
    float mean = sbuf[0] / (float)D;
    __syncthreads();

    float vv = 0.0f;
    for (int i = tid; i < D; i += 256) { float d = xr[i] - mean; vv += d * d; }
    sbuf[tid] = vv; __syncthreads();
    for (int st = 128; st > 0; st >>= 1) {
        if (tid < st) sbuf[tid] += sbuf[tid + st];
        __syncthreads();
    }
    float rstd = rsqrtf(sbuf[0] / (float)D + 1e-5f);

    for (int i = tid; i < D; i += 256) {
        float y = (xr[i] - mean) * rstd * g[i] + b[i];
        if (RELU_RES) y = res[(size_t)row * D + i] + fmaxf(y, 0.0f);
        out[(size_t)row * D + i] = y;
    }
}

// ---------------------------------------------------------------------------
// Edge-weighted scatter-add: agg[dst[e]] += x[src[e]] * w[e]   (GNN=256)
// One thread = one edge x 4-float feature group.
// ---------------------------------------------------------------------------
__global__ __launch_bounds__(256)
void edge_scatter_kernel(const int* __restrict__ src, const int* __restrict__ dst,
                         const float* __restrict__ w, const float* __restrict__ x,
                         float* __restrict__ agg, int E)
{
    long long t = (long long)blockIdx.x * 256 + threadIdx.x;
    long long total = (long long)E * 64;          // 256 feats / 4
    if (t >= total) return;
    int e = (int)(t >> 6);
    int f = (int)(t & 63) * 4;
    float wt = w[e];
    const float4 xv = *(const float4*)(x + (size_t)src[e] * 256 + f);
    float* ad = agg + (size_t)dst[e] * 256 + f;
    __hip_atomic_fetch_add(ad + 0, xv.x * wt, __ATOMIC_RELAXED, __HIP_MEMORY_SCOPE_AGENT);
    __hip_atomic_fetch_add(ad + 1, xv.y * wt, __ATOMIC_RELAXED, __HIP_MEMORY_SCOPE_AGENT);
    __hip_atomic_fetch_add(ad + 2, xv.z * wt, __ATOMIC_RELAXED, __HIP_MEMORY_SCOPE_AGENT);
    __hip_atomic_fetch_add(ad + 3, xv.w * wt, __ATOMIC_RELAXED, __HIP_MEMORY_SCOPE_AGENT);
}

__global__ __launch_bounds__(256)
void zero_kernel(float* __restrict__ p, int n)
{
    int i = blockIdx.x * 256 + threadIdx.x;
    if (i < n) p[i] = 0.0f;
}

// OOV-aware lookup: h[b] = (idx>=0) ? node_emb[idx] : oov   (GNN=256)
__global__ __launch_bounds__(256)
void lookup_kernel(const int* __restrict__ idx, const float* __restrict__ node_emb,
                   const float* __restrict__ oov, float* __restrict__ h)
{
    int b = blockIdx.x, f = threadIdx.x;
    int id = idx[b];
    h[(size_t)b * 256 + f] = (id >= 0) ? node_emb[(size_t)id * 256 + f] : oov[f];
}

// ---------------------------------------------------------------------------
extern "C" void kernel_launch(void* const* d_in, const int* in_sizes, int n_in,
                              void* d_out, int out_size, void* d_ws, size_t ws_size,
                              hipStream_t stream) {
    (void)in_sizes; (void)n_in; (void)out_size; (void)ws_size;
    constexpr int NN = 19385, NE = 1200000, G = 256, H = 512;
    constexpr int NG = 6640, NB = 6, Bb = 256;

    const int*   node_idx = (const int*)  d_in[0];
    const int*   esrc     = (const int*)  d_in[1];
    const int*   edst     = esrc + NE;
    const float* ew       = (const float*)d_in[2];
    const float* frozen   = (const float*)d_in[3];
    const float* c6W = (const float*)d_in[4];  const float* c6b = (const float*)d_in[5];
    const float* l6g = (const float*)d_in[6];  const float* l6b = (const float*)d_in[7];
    const float* c7W = (const float*)d_in[8];  const float* c7b = (const float*)d_in[9];
    const float* l7g = (const float*)d_in[10]; const float* l7b = (const float*)d_in[11];
    const float* pW  = (const float*)d_in[12]; const float* pb  = (const float*)d_in[13];
    const float* oov = (const float*)d_in[14];
    const float* ilg = (const float*)d_in[15]; const float* ilb = (const float*)d_in[16];
    const float* piW = (const float*)d_in[17]; const float* pib = (const float*)d_in[18];
    const float* rlg = (const float*)d_in[19]; const float* rlb = (const float*)d_in[20];
    const float* rW1 = (const float*)d_in[21]; const float* rb1 = (const float*)d_in[22];
    const float* rW2 = (const float*)d_in[23]; const float* rb2 = (const float*)d_in[24];
    const float* olg = (const float*)d_in[25]; const float* olb = (const float*)d_in[26];
    const float* bW  = (const float*)d_in[27]; const float* bb  = (const float*)d_in[28];
    const float* gene = (const float*)d_in[29];
    float* out = (float*)d_out;

    // ---- workspace layout ----
    char* ws = (char*)d_ws;
    size_t o = 0;
    float* agg  = (float*)(ws + o); o += (size_t)NN * G * 4;    // ~19.9 MB
    float* ybuf = (float*)(ws + o); o += (size_t)NN * G * 4;    // conv out / node_emb
    float* xbuf = (float*)(ws + o); o += (size_t)NN * G * 4;    // running node features
    float* h0   = (float*)(ws + o); o += (size_t)Bb * G * 4;
    float* hl   = (float*)(ws + o); o += (size_t)Bb * G * 4;    // ln(h0)
    float* hbuf = (float*)(ws + o); o += (size_t)Bb * H * 4;
    float* tbuf = (float*)(ws + o); o += (size_t)Bb * H * 4;
    float* ubuf = (float*)(ws + o); o += (size_t)Bb * 4 * H * 4;
    float* pert = (float*)(ws + o); o += (size_t)Bb * 3 * H * 4;

    const int nfeat = NN * G;
    const int zblocks = (nfeat + 255) / 256;
    const long long stotal = (long long)NE * 64;
    const int sblocks = (int)((stotal + 255) / 256);

    // ================= GNN layer 6 =================
    zero_kernel<<<zblocks, 256, 0, stream>>>(agg, nfeat);
    edge_scatter_kernel<<<sblocks, 256, 0, stream>>>(esrc, edst, ew, frozen, agg, NE);
    {
        dim3 g((NN + BM - 1) / BM, (G + BN - 1) / BN);
        wmma_gemm_kernel<false, EP_BIAS><<<g, 256, 0, stream>>>(agg, c6W, c6b, ybuf, NN, G, G);
    }
    ln_kernel<true><<<NN, 256, 0, stream>>>(ybuf, l6g, l6b, frozen, xbuf, G);

    // ================= GNN layer 7 =================
    zero_kernel<<<zblocks, 256, 0, stream>>>(agg, nfeat);
    edge_scatter_kernel<<<sblocks, 256, 0, stream>>>(esrc, edst, ew, xbuf, agg, NE);
    {
        dim3 g((NN + BM - 1) / BM, (G + BN - 1) / BN);
        wmma_gemm_kernel<false, EP_BIAS><<<g, 256, 0, stream>>>(agg, c7W, c7b, ybuf, NN, G, G);
    }
    ln_kernel<true><<<NN, 256, 0, stream>>>(ybuf, l7g, l7b, xbuf, xbuf, G); // in-place residual

    // ================= post_mp: node_emb = x @ postW + b =================
    {
        dim3 g((NN + BM - 1) / BM, (G + BN - 1) / BN);
        wmma_gemm_kernel<false, EP_BIAS><<<g, 256, 0, stream>>>(xbuf, pW, pb, ybuf, NN, G, G);
    }

    // ================= head =================
    lookup_kernel<<<Bb, 256, 0, stream>>>(node_idx, ybuf, oov, h0);
    ln_kernel<false><<<Bb, 256, 0, stream>>>(h0, ilg, ilb, nullptr, hl, G);
    {
        dim3 g((Bb + BM - 1) / BM, (H + BN - 1) / BN);
        wmma_gemm_kernel<false, EP_BIAS><<<g, 256, 0, stream>>>(hl, piW, pib, hbuf, Bb, H, G);
    }
    for (int i = 0; i < NB; ++i) {
        ln_kernel<false><<<Bb, 256, 0, stream>>>(hbuf, rlg + (size_t)i * H, rlb + (size_t)i * H,
                                                 nullptr, tbuf, H);
        dim3 g1((Bb + BM - 1) / BM, (4 * H + BN - 1) / BN);
        wmma_gemm_kernel<false, EP_BIAS | EP_GELU><<<g1, 256, 0, stream>>>(
            tbuf, rW1 + (size_t)i * H * 4 * H, rb1 + (size_t)i * 4 * H, ubuf, Bb, 4 * H, H);
        dim3 g2((Bb + BM - 1) / BM, (H + BN - 1) / BN);
        wmma_gemm_kernel<false, EP_BIAS | EP_ACCUM><<<g2, 256, 0, stream>>>(
            ubuf, rW2 + (size_t)i * 4 * H * H, rb2 + (size_t)i * H, hbuf, Bb, H, 4 * H);
    }
    ln_kernel<false><<<Bb, 256, 0, stream>>>(hbuf, olg, olb, nullptr, tbuf, H);
    {
        dim3 g((Bb + BM - 1) / BM, (3 * H + BN - 1) / BN);
        wmma_gemm_kernel<false, EP_BIAS><<<g, 256, 0, stream>>>(tbuf, bW, bb, pert, Bb, 3 * H, H);
    }
    // logits[768,6640] = pert[768,512] @ gene[6640,512]^T   (NT gemm, straight to d_out)
    {
        dim3 g((Bb * 3 + BM - 1) / BM, (NG + BN - 1) / BN);
        wmma_gemm_kernel<true, EP_NONE><<<g, 256, 0, stream>>>(pert, gene, nullptr, out,
                                                              Bb * 3, NG, H);
    }
}
